// BioGuardGAT_5798205849901
// MI455X (gfx1250) — matrix operand order repaired and checked by reading, not compiled, not run on hardware
//
#include <hip/hip_runtime.h>
#include <hip/hip_bf16.h>

typedef __attribute__((ext_vector_type(16))) __bf16 v16bf;
typedef __attribute__((ext_vector_type(8)))  float  v8f;

// ---------- order-preserving float <-> uint for atomic max ----------
__device__ __forceinline__ unsigned ford(float f) {
    unsigned u = __float_as_uint(f);
    return (u & 0x80000000u) ? ~u : (u | 0x80000000u);
}
__device__ __forceinline__ float funord(unsigned o) {
    unsigned u = (o & 0x80000000u) ? (o & 0x7fffffffu) : ~o;
    return __uint_as_float(u);
}

// ---------- fills ----------
__global__ void k_fill_f32(float* p, float v, int n) {
    int i = blockIdx.x * blockDim.x + threadIdx.x;
    if (i < n) p[i] = v;
}
__global__ void k_fill_u32(unsigned* p, unsigned v, int n) {
    int i = blockIdx.x * blockDim.x + threadIdx.x;
    if (i < n) p[i] = v;
}

// ---------- WMMA bf16 GEMM: out[M,NOUT] = act(bn(A[M,K] @ W[K,NOUT] + bias)) ----------
// K and NOUT are compile-time so every fragment load is base + immediate offset.
// One wave computes a 16x64 C strip (4 accumulators); the A fragment is reused
// across 4 N-tiles -> 4 v_wmma per 32-wide K step.
// Fragment layouts used here (A and B share one contiguous k order):
//  A (16x32 bf16): lane L row M=L%16, group g=L/16; elem j (VGPR v=j/2):
//      k = 16*g + j
//  B (32x16 bf16): lane L col N=L%16; elem j: k = 16*g + j
//  C (16x16 f32):  lane L col N=L%16; VGPR r -> row M = r + 8*g
// A tile staged via per-wave LDS tile (coalesced global loads; stride-33 rows
// keep the fragment gather bank-conflict-free). Main loop has no guards; the
// ragged K tail (encoder K=30) is an if-constexpr block with branchless guards.
template <int K, int NOUT>
__global__ void k_gemm_bf16(const float* __restrict__ A, int M,
                            const float* __restrict__ W,
                            const float* __restrict__ bias,
                            const float* __restrict__ bng, const float* __restrict__ bnb,
                            const float* __restrict__ bnm, const float* __restrict__ bnv,
                            int act, float* __restrict__ out)
{
    __shared__ float AtAll[4 * 16 * 33];
    int tid  = threadIdx.x;
    int wave = tid >> 5;
    int lane = tid & 31;
    float* At = AtAll + wave * (16 * 33);

    constexpr int stripsN = NOUT >> 6;        // NOUT / 64
    int stripsTotal = (M >> 4) * stripsN;
    int strip = blockIdx.x * (blockDim.x >> 5) + wave;
    if (strip >= stripsTotal) return;
    int tm = strip / stripsN, tn = strip % stripsN;
    int mr = tm << 4, nc = tn << 6;
    int g = lane >> 4, l16 = lane & 15;

    constexpr int KFULL = K & ~31;

    const float* Arow = A + (size_t)mr * K + lane;                 // + r*K + k0
    const float* Wb   = W + (size_t)(g << 4) * NOUT + nc + l16;    // + (k0+j)*NOUT + nt*16
    const float* Afr  = At + l16 * 33;                             // fragment gather base

    v8f acc[4] = {};

    for (int k0 = 0; k0 < KFULL; k0 += 32) {
        // ---- stage A tile (16 rows x 32 k), coalesced, no guards ----
#pragma unroll
        for (int r = 0; r < 16; ++r)
            At[r * 33 + lane] = Arow[r * K + k0];
        // ---- A fragment from LDS ----
        v16bf a;
#pragma unroll
        for (int j = 0; j < 16; ++j) {
            int v   = j >> 1;
            int kla = (g << 4) + j + (v & 0);
            a[j] = (__bf16)Afr[kla];
        }
        // ---- 4 B fragments direct from global: one base + immediate offsets ----
        const float* Wk = Wb + (size_t)k0 * NOUT;
#pragma unroll
        for (int nt = 0; nt < 4; ++nt) {
            v16bf b;
#pragma unroll
            for (int j = 0; j < 16; ++j)
                b[j] = (__bf16)Wk[j * NOUT + nt * 16];
            acc[nt] = __builtin_amdgcn_wmma_f32_16x16x32_bf16(
                false, a, false, b, (short)0, acc[nt], false, false);
        }
    }

    if constexpr ((K & 31) != 0) {   // ragged tail (encoder K=30), branchless guards
        constexpr int k0 = KFULL;
        int kk = k0 + lane;
        int kc = (kk < K) ? kk : (K - 1);
#pragma unroll
        for (int r = 0; r < 16; ++r) {
            float av = A[(size_t)(mr + r) * K + kc];
            At[r * 33 + lane] = (kk < K) ? av : 0.0f;
        }
        v16bf a;
#pragma unroll
        for (int j = 0; j < 16; ++j) {
            int v   = j >> 1;
            int kla = (g << 4) + j + (v & 0);
            a[j] = (__bf16)Afr[kla];
        }
#pragma unroll
        for (int nt = 0; nt < 4; ++nt) {
            v16bf b;
#pragma unroll
            for (int j = 0; j < 16; ++j) {
                int kb  = k0 + (g << 4) + j;
                int kbc = (kb < K) ? kb : (K - 1);
                float bv = W[(size_t)kbc * NOUT + nc + nt * 16 + l16];
                b[j] = (__bf16)((kb < K) ? bv : 0.0f);
            }
            acc[nt] = __builtin_amdgcn_wmma_f32_16x16x32_bf16(
                false, a, false, b, (short)0, acc[nt], false, false);
        }
    }

    // ---- epilogue: bias, folded BatchNorm, activation ----
#pragma unroll
    for (int nt = 0; nt < 4; ++nt) {
        int col = nc + nt * 16 + l16;
        float bcol = bias ? bias[col] : 0.0f;
        float s = 1.0f, t = 0.0f;
        if (bng) {
            float inv = rsqrtf(bnv[col] + 1e-5f);
            s = bng[col] * inv;
            t = bnb[col] - bnm[col] * s;
        }
#pragma unroll
        for (int r = 0; r < 8; ++r) {
            int row = mr + r + (g << 3);
            float v = (acc[nt][r] + bcol) * s + t;
            if (act == 1) v = fmaxf(v, 0.0f);
            out[(size_t)row * NOUT + col] = v;
        }
    }
}

// ---------- mean of edge_attr over E (for conv1 self-loop fill) ----------
__global__ void k_ea_mean(const float* __restrict__ ea, int E, float* __restrict__ loopvec)
{
    int k    = threadIdx.x >> 5;   // 0..10
    int lane = threadIdx.x & 31;
    if (k >= 11) return;
    float acc = 0.0f;
    for (int e = lane; e < E; e += 32) acc += ea[(size_t)e * 11 + k];
    for (int off = 16; off; off >>= 1) acc += __shfl_xor(acc, off, 32);
    if (lane == 0) loopvec[k] = acc / (float)E;
}

// ---------- GATv2 attention logit + segment max (wave per (edge, head)) ----------
__global__ void k_edge_logit(const float* __restrict__ xl, const float* __restrict__ xr,
                             const int* __restrict__ src, const int* __restrict__ dst,
                             const float* __restrict__ ea, const float* __restrict__ We,
                             const float* __restrict__ att, const float* __restrict__ loopvec,
                             int E, int E2, int H,
                             float* __restrict__ logit, unsigned* __restrict__ nmax)
{
    int w    = blockIdx.x * (blockDim.x >> 5) + (threadIdx.x >> 5);
    int lane = threadIdx.x & 31;
    if (w >= E2 * H) return;
    int e = w / H, h = w % H;
    int s, d;
    if (e < E) { s = src[e]; d = dst[e]; } else { s = e - E; d = s; }

    float eak[11];
#pragma unroll
    for (int k = 0; k < 11; ++k)
        eak[k] = (e < E) ? ea[(size_t)e * 11 + k] : (loopvec ? loopvec[k] : 0.0f);

    int HC = H * 128;
    const float* xls = xl + (size_t)s * HC + h * 128;
    const float* xrd = xr + (size_t)d * HC + h * 128;
    const float* wh  = We + h * 128;
    const float* ah  = att + h * 128;

    float sum = 0.0f;
#pragma unroll
    for (int cc = 0; cc < 4; ++cc) {
        int cidx = lane + cc * 32;
        float v = xls[cidx] + xrd[cidx];
#pragma unroll
        for (int k = 0; k < 11; ++k) v += eak[k] * wh[(size_t)k * HC + cidx];
        v = (v > 0.0f) ? v : 0.2f * v;          // LeakyReLU(0.2)
        sum += v * ah[cidx];
    }
    for (int off = 16; off; off >>= 1) sum += __shfl_xor(sum, off, 32);
    if (lane == 0) {
        logit[w] = sum;
        atomicMax(&nmax[(size_t)d * H + h], ford(sum));
    }
}

// ---------- p = exp(logit - max); segment sum ----------
__global__ void k_edge_p(const int* __restrict__ dst, int E, int E2, int H,
                         const unsigned* __restrict__ nmax, float* __restrict__ logit,
                         float* __restrict__ nsum)
{
    int i = blockIdx.x * blockDim.x + threadIdx.x;
    if (i >= E2 * H) return;
    int e = i / H, h = i % H;
    int d = (e < E) ? dst[e] : (e - E);
    float p = __expf(logit[i] - funord(nmax[(size_t)d * H + h]));
    logit[i] = p;
    atomicAdd(&nsum[(size_t)d * H + h], p);
}

// ---------- out[d] += alpha * xl[s] ----------
__global__ void k_edge_acc(const int* __restrict__ src, const int* __restrict__ dst,
                           int E, int E2, int H,
                           const float* __restrict__ p, const float* __restrict__ nsum,
                           const float* __restrict__ xl, float* __restrict__ out)
{
    int HC = H * 128;
    long long total = (long long)E2 * HC;
    long long i = (long long)blockIdx.x * blockDim.x + threadIdx.x;
    if (i >= total) return;
    int e = (int)(i / HC);
    int r = (int)(i % HC);
    int h = r >> 7;
    int s, d;
    if (e < E) { s = src[e]; d = dst[e]; } else { s = e - E; d = s; }
    float alpha = p[(size_t)e * H + h] / nsum[(size_t)d * H + h];
    atomicAdd(&out[(size_t)d * HC + r], alpha * xl[(size_t)s * HC + r]);
}

// ---------- out = ELU(out + bias) ----------
__global__ void k_bias_elu(float* __restrict__ out, const float* __restrict__ bias,
                           int HC, long long total)
{
    long long i = (long long)blockIdx.x * blockDim.x + threadIdx.x;
    if (i >= total) return;
    int c = (int)(i % HC);
    float v = out[i] + bias[c];
    out[i] = (v > 0.0f) ? v : (__expf(v) - 1.0f);
}

// ---------- graph pooling (mean + max over batch segments) ----------
__global__ void k_pool_acc(const float* __restrict__ h, const int* __restrict__ batch,
                           int N, float* __restrict__ psum, unsigned* __restrict__ pmax,
                           float* __restrict__ cnt)
{
    int i = blockIdx.x * blockDim.x + threadIdx.x;
    if (i >= N * 128) return;
    int n = i >> 7, c = i & 127;
    int b = batch[n];
    float v = h[i];
    atomicAdd(&psum[b * 128 + c], v);
    atomicMax(&pmax[b * 128 + c], ford(v));
    if (c == 0) atomicAdd(&cnt[b], 1.0f);
}
__global__ void k_pool_fin(const float* __restrict__ psum, const unsigned* __restrict__ pmax,
                           const float* __restrict__ cnt, int B, float* __restrict__ va)
{
    int i = blockIdx.x * blockDim.x + threadIdx.x;
    if (i >= B * 128) return;
    int b = i >> 7, c = i & 127;
    va[b * 256 + c]       = psum[i] / cnt[b];
    va[b * 256 + 128 + c] = funord(pmax[i]);
}

// ---------- head MLPs: one 256-thread block per graph ----------
__global__ void k_heads(const float* __restrict__ va, const float* __restrict__ vb,
                        const float* __restrict__ enza, const float* __restrict__ enzb,
                        const float* g1W, const float* g1b, const float* g1g, const float* g1be,
                        const float* g1rm, const float* g1rv,
                        const float* g2W, const float* g2b,
                        const float* g3W, const float* g3b,
                        const float* p1W, const float* p1b,
                        const float* p2W, const float* p2b,
                        const float* a1W, const float* a1b,
                        const float* a2W, const float* a2b,
                        float* __restrict__ outp)
{
    __shared__ float gat[768], enz[45], h1[256], h2[64], q[64], av[32];
    int b = blockIdx.x, tid = threadIdx.x;
    for (int j = tid; j < 768; j += blockDim.x) {
        int part = j >> 8, c = j & 255;
        float x = va[b * 256 + c], y = vb[b * 256 + c];
        gat[j] = (part == 0) ? (x + y) : (part == 1 ? fabsf(x - y) : x * y);
    }
    for (int j = tid; j < 45; j += blockDim.x) {
        int part = j / 15, k = j % 15;
        float x = enza[b * 15 + k], y = enzb[b * 15 + k];
        enz[j] = (part == 0) ? (x + y) : (part == 1 ? fabsf(x - y) : x * y);
    }
    __syncthreads();
    {   // g1: 768 -> 256, BN + ReLU
        int j = tid;
        float acc = g1b[j];
        for (int i = 0; i < 768; ++i) acc += gat[i] * g1W[i * 256 + j];
        float s = g1g[j] * rsqrtf(g1rv[j] + 1e-5f);
        acc = (acc - g1rm[j]) * s + g1be[j];
        h1[j] = fmaxf(acc, 0.0f);
    }
    __syncthreads();
    if (tid < 64) {                       // g2: 256 -> 64
        float acc = g2b[tid];
        for (int i = 0; i < 256; ++i) acc += h1[i] * g2W[i * 64 + tid];
        h2[tid] = fmaxf(acc, 0.0f);
    } else if (tid < 128) {               // p1: 45 -> 64
        int j = tid - 64;
        float acc = p1b[j];
        for (int i = 0; i < 45; ++i) acc += enz[i] * p1W[i * 64 + j];
        q[j] = fmaxf(acc, 0.0f);
    } else if (tid < 160) {               // a1: 813 -> 32
        int j = tid - 128;
        float acc = a1b[j];
        for (int i = 0; i < 768; ++i) acc += gat[i] * a1W[i * 32 + j];
        for (int i = 0; i < 45; ++i)  acc += enz[i] * a1W[(768 + i) * 32 + j];
        av[j] = fmaxf(acc, 0.0f);
    }
    __syncthreads();
    if (tid == 0) {
        float gl = g3b[0];
        for (int i = 0; i < 64; ++i) gl += h2[i] * g3W[i];
        float pr = p2b[0];
        for (int i = 0; i < 64; ++i) pr += q[i] * p2W[i];
        float aa = a2b[0];
        for (int i = 0; i < 32; ++i) aa += av[i] * a2W[i];
        float alpha = 1.0f / (1.0f + __expf(-aa));
        outp[b] = alpha * gl + (1.0f - alpha) * pr;
    }
}

// =====================================================================
extern "C" void kernel_launch(void* const* d_in, const int* in_sizes, int n_in,
                              void* d_out, int out_size, void* d_ws, size_t ws_size,
                              hipStream_t stream)
{
    const int N = in_sizes[0] / 30;
    const int E = in_sizes[1] / 2;
    const int B = in_sizes[4] / 15;
    const int E2 = E + N;

    const float* x_a     = (const float*)d_in[0];
    const int*   ei_a    = (const int*)d_in[1];
    const float* ea_a    = (const float*)d_in[2];
    const int*   batch_a = (const int*)d_in[3];
    const float* enz_a   = (const float*)d_in[4];
    const float* x_b     = (const float*)d_in[5];
    const int*   ei_b    = (const int*)d_in[6];
    const float* ea_b    = (const float*)d_in[7];
    const int*   batch_b = (const int*)d_in[8];
    const float* enz_b   = (const float*)d_in[9];
    const float* P[38];
    for (int i = 0; i < 38; ++i) P[i] = (const float*)d_in[10 + i];
    const float *enc_W=P[0], *enc_b=P[1], *enc_g=P[2], *enc_be=P[3], *enc_rm=P[4], *enc_rv=P[5];
    const float *c1_Wl=P[6], *c1_bl=P[7], *c1_Wr=P[8], *c1_br=P[9], *c1_We=P[10], *c1_att=P[11], *c1_bias=P[12];
    const float *c2_Wl=P[13], *c2_bl=P[14], *c2_Wr=P[15], *c2_br=P[16], *c2_We=P[17], *c2_att=P[18], *c2_bias=P[19];
    const float *g1_W=P[20], *g1_b=P[21], *g1_g=P[22], *g1_be=P[23], *g1_rm=P[24], *g1_rv=P[25];
    const float *g2_W=P[26], *g2_b=P[27], *g3_W=P[28], *g3_b=P[29];
    const float *p1_W=P[30], *p1_b=P[31], *p2_W=P[32], *p2_b=P[33];
    const float *a1_W=P[34], *a1_b=P[35], *a2_W=P[36], *a2_b=P[37];

    // ---- workspace carve (256B aligned) ----
    char* w = (char*)d_ws;
    auto carve = [&](size_t bytes) { char* p = w; w += (bytes + 255) & ~(size_t)255; return p; };
    float*    h0      = (float*)carve((size_t)N * 128 * 4);
    float*    xl1     = (float*)carve((size_t)N * 512 * 4);
    float*    xr1     = (float*)carve((size_t)N * 512 * 4);
    float*    out1    = (float*)carve((size_t)N * 512 * 4);
    float*    xl2     = (float*)carve((size_t)N * 128 * 4);
    float*    xr2     = (float*)carve((size_t)N * 128 * 4);
    float*    out2    = (float*)carve((size_t)N * 128 * 4);
    float*    logit   = (float*)carve((size_t)E2 * 4 * 4);
    unsigned* nmax    = (unsigned*)carve((size_t)N * 4 * 4);
    float*    nsum    = (float*)carve((size_t)N * 4 * 4);
    float*    psum    = (float*)carve((size_t)B * 128 * 4);
    unsigned* pmax    = (unsigned*)carve((size_t)B * 128 * 4);
    float*    cnt     = (float*)carve((size_t)B * 4);
    float*    va      = (float*)carve((size_t)B * 256 * 4);
    float*    vb      = (float*)carve((size_t)B * 256 * 4);
    float*    loopvec = (float*)carve(64);

    auto fillF = [&](float* p, float v, long long n) {
        k_fill_f32<<<dim3((unsigned)((n + 255) / 256)), dim3(256), 0, stream>>>(p, v, (int)n);
    };
    auto fillU = [&](unsigned* p, unsigned v, long long n) {
        k_fill_u32<<<dim3((unsigned)((n + 255) / 256)), dim3(256), 0, stream>>>(p, v, (int)n);
    };
    auto grid_strips = [&](int M, int Nout) {
        int strips = (M / 16) * (Nout / 64);   // one wave per 16x64 strip
        return dim3((strips + 3) / 4);
    };

    auto run_arm = [&](const float* x, const int* ei, const float* ea,
                       const int* batch, float* vout) {
        const int* src = ei;
        const int* dst = ei + E;

        // encoder: [N,30] @ [30,128], BN + ReLU
        k_gemm_bf16<30, 128><<<grid_strips(N, 128), dim3(128), 0, stream>>>(
            x, N, enc_W, enc_b, enc_g, enc_be, enc_rm, enc_rv, 1, h0);

        // ---- conv1 (H=4, C=128) ----
        k_gemm_bf16<128, 512><<<grid_strips(N, 512), dim3(128), 0, stream>>>(
            h0, N, c1_Wl, c1_bl, nullptr, nullptr, nullptr, nullptr, 0, xl1);
        k_gemm_bf16<128, 512><<<grid_strips(N, 512), dim3(128), 0, stream>>>(
            h0, N, c1_Wr, c1_br, nullptr, nullptr, nullptr, nullptr, 0, xr1);
        k_ea_mean<<<dim3(1), dim3(352), 0, stream>>>(ea, E, loopvec);
        fillU(nmax, 0u, (long long)N * 4);
        fillF(nsum, 0.0f, (long long)N * 4);
        fillF(out1, 0.0f, (long long)N * 512);
        {
            int waves = E2 * 4;
            k_edge_logit<<<dim3((waves + 3) / 4), dim3(128), 0, stream>>>(
                xl1, xr1, src, dst, ea, c1_We, c1_att, loopvec, E, E2, 4, logit, nmax);
            int tot = E2 * 4;
            k_edge_p<<<dim3((tot + 255) / 256), dim3(256), 0, stream>>>(
                dst, E, E2, 4, nmax, logit, nsum);
            long long tot2 = (long long)E2 * 512;
            k_edge_acc<<<dim3((unsigned)((tot2 + 255) / 256)), dim3(256), 0, stream>>>(
                src, dst, E, E2, 4, logit, nsum, xl1, out1);
            long long tot3 = (long long)N * 512;
            k_bias_elu<<<dim3((unsigned)((tot3 + 255) / 256)), dim3(256), 0, stream>>>(
                out1, c1_bias, 512, tot3);
        }

        // ---- conv2 (H=1, C=128) ----
        k_gemm_bf16<512, 128><<<grid_strips(N, 128), dim3(128), 0, stream>>>(
            out1, N, c2_Wl, c2_bl, nullptr, nullptr, nullptr, nullptr, 0, xl2);
        k_gemm_bf16<512, 128><<<grid_strips(N, 128), dim3(128), 0, stream>>>(
            out1, N, c2_Wr, c2_br, nullptr, nullptr, nullptr, nullptr, 0, xr2);
        fillU(nmax, 0u, N);
        fillF(nsum, 0.0f, N);
        fillF(out2, 0.0f, (long long)N * 128);
        {
            int waves = E2;
            k_edge_logit<<<dim3((waves + 3) / 4), dim3(128), 0, stream>>>(
                xl2, xr2, src, dst, ea, c2_We, c2_att, nullptr, E, E2, 1, logit, nmax);
            k_edge_p<<<dim3((E2 + 255) / 256), dim3(256), 0, stream>>>(
                dst, E, E2, 1, nmax, logit, nsum);
            long long tot2 = (long long)E2 * 128;
            k_edge_acc<<<dim3((unsigned)((tot2 + 255) / 256)), dim3(256), 0, stream>>>(
                src, dst, E, E2, 1, logit, nsum, xl2, out2);
            long long tot3 = (long long)N * 128;
            k_bias_elu<<<dim3((unsigned)((tot3 + 255) / 256)), dim3(256), 0, stream>>>(
                out2, c2_bias, 128, tot3);
        }

        // ---- pooling ----
        fillF(psum, 0.0f, (long long)B * 128);
        fillU(pmax, 0u, (long long)B * 128);
        fillF(cnt, 0.0f, B);
        k_pool_acc<<<dim3((N * 128 + 255) / 256), dim3(256), 0, stream>>>(
            out2, batch, N, psum, pmax, cnt);
        k_pool_fin<<<dim3((B * 128 + 255) / 256), dim3(256), 0, stream>>>(
            psum, pmax, cnt, B, vout);
    };

    run_arm(x_a, ei_a, ea_a, batch_a, va);
    run_arm(x_b, ei_b, ea_b, batch_b, vb);

    k_heads<<<dim3(B), dim3(256), 0, stream>>>(
        va, vb, enz_a, enz_b,
        g1_W, g1_b, g1_g, g1_be, g1_rm, g1_rv,
        g2_W, g2_b, g3_W, g3_b,
        p1_W, p1_b, p2_W, p2_b,
        a1_W, a1_b, a2_W, a2_b,
        (float*)d_out);
}
